// GCN_local_54039278518881
// MI455X (gfx1250) — matrix operand order repaired and checked by reading, not compiled, not run on hardware
//
#include <hip/hip_runtime.h>
#include <hip/hip_bf16.h>

typedef __attribute__((ext_vector_type(16))) __bf16    v16bf;
typedef __attribute__((ext_vector_type(2)))  __bf16    bf16x2;
typedef __attribute__((ext_vector_type(8)))  float     v8f;
typedef __attribute__((ext_vector_type(8)))  unsigned  v8u;
typedef __attribute__((ext_vector_type(4)))  unsigned  u32x4;
typedef __attribute__((ext_vector_type(4)))  float     f32x4;
typedef __attribute__((ext_vector_type(2)))  float     f32x2;

// ---------- helpers ----------
// Let the backend pick the packed HW conversion (v_cvt_pk_bf16_f32 on CDNA5).
__device__ inline unsigned packbf2(float lo, float hi) {
    f32x2 v; v[0] = lo; v[1] = hi;
    return __builtin_bit_cast(unsigned, __builtin_convertvector(v, bf16x2));
}

// ---------- stage 1: h0 += x @ W1  (M=N, N=32, K split over blockIdx.y) ------
// 128 threads = 4 waves; each wave owns one 16x16 tile of a 32x32 output
// block. K-loop step 32, double-buffered LDS of packed bf16 K-pairs; the
// guarded K-tail is peeled out of the steady-state loop.
#define KSPLIT 8
__global__ __launch_bounds__(128) void gemm_h0_wmma(
    const float* __restrict__ x, const float* __restrict__ W1,
    float* __restrict__ h0, int N, int K) {
    __shared__ __attribute__((aligned(16))) unsigned lAp[2][32][20];
    __shared__ __attribute__((aligned(16))) unsigned lBp[2][16][36];

    const int tid  = threadIdx.x;
    const int wave = tid >> 5;
    const int lane = tid & 31;
    const int m0   = blockIdx.x * 32;
    const int mt   = (wave >> 1) * 16;
    const int nt   = (wave & 1) * 16;
    const int hh   = lane >> 4;
    const int ml   = lane & 15;
    const int arow = mt + ml;

    const int r     = tid >> 2;            // A row 0..31
    const int pb    = (tid & 3) * 4;       // A pair base 0,4,8,12
    const int cbase = pb * 2;              // A float base 0,8,16,24
    const int pr    = tid >> 3;            // B pair-row 0..15
    const int cb    = (tid & 7) * 4;       // B col base 0..28

    const int KC     = (((K + KSPLIT * 32 - 1) / (KSPLIT * 32)) * 32);
    const int kstart = blockIdx.y * KC;
    if (kstart >= K) return;
    const int kend    = min(kstart + KC, K);
    const int kcount  = kend - kstart;
    const int nfull   = kcount >> 5;             // complete 32-wide tiles
    const bool has_tail = (nfull << 5) < kcount; // last chunk only

    const int xrow = min(m0 + r, N - 1);
    const float* __restrict__ xbase = x + (size_t)xrow * K;

    f32x2 ra[4];
    f32x4 rb0, rb1;

    auto load_full = [&](int kk) {               // unconditional vector loads
        const f32x2* ap = (const f32x2*)(xbase + kk + cbase);
#pragma unroll
        for (int j = 0; j < 4; ++j)
            ra[j] = __builtin_nontemporal_load(ap + j);   // x is single-use
        const int k0 = kk + 2 * pr;
        rb0 = *(const f32x4*)(W1 + (size_t)k0 * 32 + cb);
        rb1 = *(const f32x4*)(W1 + (size_t)(k0 + 1) * 32 + cb);
    };
    auto load_tail = [&](int kk) {               // guarded, executes once
#pragma unroll
        for (int j = 0; j < 4; ++j) {
            int g0 = kk + cbase + 2 * j, g1 = g0 + 1;
            ra[j][0] = (g0 < K) ? xbase[g0] : 0.0f;
            ra[j][1] = (g1 < K) ? xbase[g1] : 0.0f;
        }
        const int k0 = kk + 2 * pr;
#pragma unroll
        for (int j = 0; j < 4; ++j) {
            rb0[j] = (k0     < K) ? W1[(size_t)k0 * 32 + cb + j]       : 0.0f;
            rb1[j] = (k0 + 1 < K) ? W1[(size_t)(k0 + 1) * 32 + cb + j] : 0.0f;
        }
    };
    auto store_lds = [&](int buf) {
        u32x4 ua, ub;
#pragma unroll
        for (int j = 0; j < 4; ++j) {
            ua[j] = packbf2(ra[j][0], ra[j][1]);
            ub[j] = packbf2(rb0[j], rb1[j]);
        }
        *(u32x4*)&lAp[buf][r][pb]  = ua;
        *(u32x4*)&lBp[buf][pr][cb] = ub;
    };

    v8f acc = {};
    auto compute = [&](int buf) {
        v8u au, bu;
#pragma unroll
        for (int v = 0; v < 4; ++v) au[v] = lAp[buf][arow][4 * hh + v];
#pragma unroll
        for (int v = 0; v < 4; ++v) au[4 + v] = lAp[buf][arow][8 + 4 * hh + v];
#pragma unroll
        for (int v = 0; v < 8; ++v) bu[v] = lBp[buf][(v & 3) + 4 * hh + 8 * (v >> 2)][nt + ml];
        acc = __builtin_amdgcn_wmma_f32_16x16x32_bf16(
            false, __builtin_bit_cast(v16bf, au), false,
            __builtin_bit_cast(v16bf, bu), (short)0, acc, false, false);
    };

    int cur = 0;
    if (nfull > 0) {
        load_full(kstart);
        store_lds(0);
        __syncthreads();
        // steady state: always a next full tile; no guards anywhere
        for (int i = 0; i + 1 < nfull; ++i) {
            load_full(kstart + (i + 1) * 32);
            compute(cur);
            store_lds(cur ^ 1);       // buf cur^1 last read before the previous sync
            __syncthreads();
            cur ^= 1;
        }
        // last full tile (+ optional peeled tail)
        if (has_tail) load_tail(kstart + nfull * 32);
        compute(cur);
        if (has_tail) {
            store_lds(cur ^ 1);
            __syncthreads();
            compute(cur ^ 1);
        }
    } else if (has_tail) {            // tiny chunk: tail only
        load_tail(kstart);
        store_lds(0);
        __syncthreads();
        compute(0);
    }

#pragma unroll
    for (int v = 0; v < 8; ++v) {
        const int gm = m0 + mt + v + 8 * hh;
        if (gm < N) unsafeAtomicAdd(&h0[(size_t)gm * 32 + nt + ml], acc[v]);
    }
}

__global__ void zero_buf(float* __restrict__ p, int n) {
    int i = blockIdx.x * blockDim.x + threadIdx.x;
    if (i < n) p[i] = 0.0f;
}

// ---------- stage 3: hw2 = h @ W2  (K=32, N=16), one WMMA per 16-row tile ----
__global__ __launch_bounds__(256) void gemm_hw2_wmma(
    const float* __restrict__ h, const float* __restrict__ W2,
    float* __restrict__ hw2, int N) {
    __shared__ unsigned lBp[16][17];         // packed K-pairs of W2 (32x16)
    const int tid = threadIdx.x;
    if (tid < 256) {
        int prr = tid >> 4, c = tid & 15;
        lBp[prr][c] = packbf2(W2[(size_t)(2 * prr) * 16 + c],
                              W2[(size_t)(2 * prr + 1) * 16 + c]);
    }
    __syncthreads();

    const int wave = tid >> 5, lane = tid & 31;
    const int tile = blockIdx.x * 8 + wave;
    const int m0   = tile * 16;
    if (m0 >= N) return;                     // whole wave; no more barriers

    const int hh = lane >> 4, ml = lane & 15;
    int row = m0 + ml; if (row >= N) row = N - 1;
    const f32x2* hr = (const f32x2*)(h + (size_t)row * 32);

    v8u au, bu;
#pragma unroll
    for (int v = 0; v < 4; ++v) {
        f32x2 p0 = hr[4 * hh + v];           // K = 8hh+2v, +1
        f32x2 p1 = hr[8 + 4 * hh + v];       // K = 16+8hh+2v, +1
        au[v]     = packbf2(p0[0], p0[1]);
        au[4 + v] = packbf2(p1[0], p1[1]);
    }
#pragma unroll
    for (int v = 0; v < 8; ++v) bu[v] = lBp[(v & 3) + 4 * hh + 8 * (v >> 2)][ml];

    v8f c = {};
    c = __builtin_amdgcn_wmma_f32_16x16x32_bf16(
        false, __builtin_bit_cast(v16bf, au), false,
        __builtin_bit_cast(v16bf, bu), (short)0, c, false, false);
#pragma unroll
    for (int v = 0; v < 8; ++v) {
        const int gm = m0 + v + 8 * hh;
        if (gm < N) hw2[(size_t)gm * 16 + ml] = c[v];
    }
}

// ---------- sparse / elementwise kernels ----------
__global__ void deg_init(float* __restrict__ deg, int N) {
    int i = blockIdx.x * blockDim.x + threadIdx.x;
    if (i < N) deg[i] = 1.0f;
}

__global__ void deg_scatter(const int* __restrict__ ei1,
                            const float* __restrict__ ew,
                            float* __restrict__ deg, int E) {
    int e = blockIdx.x * blockDim.x + threadIdx.x;
    if (e < E) unsafeAtomicAdd(&deg[ei1[e]], ew[e]);
}

__global__ void dinv_diag1(const float* __restrict__ deg,
                           float* __restrict__ dinv,
                           const float* __restrict__ h0,
                           float* __restrict__ agg1, int N) {
    int t = blockIdx.x * blockDim.x + threadIdx.x;
    if (t >= N * 32) return;
    int i = t >> 5;
    float r = rsqrtf(deg[i]);
    if ((t & 31) == 0) dinv[i] = r;
    agg1[t] = r * r * h0[t];
}

__global__ void scatter1(const int* __restrict__ ei0, const int* __restrict__ ei1,
                         const float* __restrict__ ew, const float* __restrict__ dinv,
                         const float* __restrict__ h0, float* __restrict__ agg1, int E) {
    long t = (long)blockIdx.x * blockDim.x + threadIdx.x;
    if (t >= (long)E * 32) return;
    int e = (int)(t >> 5), c = (int)(t & 31);
    int s = ei0[e], d = ei1[e];
    float coef = ew[e] * dinv[s] * dinv[d];
    unsafeAtomicAdd(&agg1[(size_t)d * 32 + c], coef * h0[(size_t)s * 32 + c]);
}

__global__ void relu_bias(float* __restrict__ agg1, const float* __restrict__ b1, int N) {
    int t = blockIdx.x * blockDim.x + threadIdx.x;
    if (t < N * 32) agg1[t] = fmaxf(agg1[t] + b1[t & 31], 0.0f);
}

__global__ void dinv_diag2(const float* __restrict__ dinv,
                           const float* __restrict__ hw2,
                           float* __restrict__ agg2, int N) {
    int t = blockIdx.x * blockDim.x + threadIdx.x;
    if (t >= N * 16) return;
    float r = dinv[t >> 4];
    agg2[t] = r * r * hw2[t];
}

__global__ void scatter2(const int* __restrict__ ei0, const int* __restrict__ ei1,
                         const float* __restrict__ ew, const float* __restrict__ dinv,
                         const float* __restrict__ hw2, float* __restrict__ agg2, int E) {
    long t = (long)blockIdx.x * blockDim.x + threadIdx.x;
    if (t >= (long)E * 16) return;
    int e = (int)(t >> 4), c = (int)(t & 15);
    int s = ei0[e], d = ei1[e];
    float coef = ew[e] * dinv[s] * dinv[d];
    unsafeAtomicAdd(&agg2[(size_t)d * 16 + c], coef * hw2[(size_t)s * 16 + c]);
}

__global__ void tanh_bias(const float* __restrict__ agg2, const float* __restrict__ b2,
                          float* __restrict__ z, int N) {
    int t = blockIdx.x * blockDim.x + threadIdx.x;
    if (t < N * 16) z[t] = tanhf(agg2[t] + b2[t & 15]);
}

// ---------- launch ----------
extern "C" void kernel_launch(void* const* d_in, const int* in_sizes, int n_in,
                              void* d_out, int out_size, void* d_ws, size_t ws_size,
                              hipStream_t stream) {
    const float* x  = (const float*)d_in[0];
    const int*   ei = (const int*)  d_in[1];   // [T,2,E]
    const float* ew = (const float*)d_in[2];   // [T,E]
    const float* W1 = (const float*)d_in[3];   // [IN_DIM,32]
    const float* b1 = (const float*)d_in[4];   // [32]
    const float* W2 = (const float*)d_in[5];   // [32,16]
    const float* b2 = (const float*)d_in[6];   // [16]
    float* out = (float*)d_out;

    const int H      = in_sizes[4];            // 32
    const int IN_DIM = in_sizes[3] / H;
    const int N      = in_sizes[0] / IN_DIM;
    const int Z      = in_sizes[5] / H;        // 16
    const int T      = out_size / (N * Z);
    const int E      = in_sizes[2] / T;

    float* ws   = (float*)d_ws;
    float* h0   = ws;                              // N*32
    float* deg  = h0   + (size_t)N * 32;           // N
    float* dinv = deg  + N;                        // N
    float* agg1 = dinv + N;                        // N*32
    float* hw2  = agg1 + (size_t)N * 32;           // N*16
    float* agg2 = hw2  + (size_t)N * 16;           // N*16

    const int thr = 256;

    // stage 1 (once): h0 = x @ W1 via bf16 WMMA, K split across blockIdx.y
    zero_buf<<<(N * 32 + thr - 1) / thr, thr, 0, stream>>>(h0, N * 32);
    dim3 g1((N + 31) / 32, KSPLIT);
    gemm_h0_wmma<<<g1, 128, 0, stream>>>(x, W1, h0, N, IN_DIM);

    for (int t = 0; t < T; ++t) {
        const int*   ei0 = ei + (size_t)t * 2 * E;
        const int*   ei1 = ei0 + E;
        const float* ewt = ew + (size_t)t * E;

        deg_init   <<<(N + thr - 1) / thr, thr, 0, stream>>>(deg, N);
        deg_scatter<<<(E + thr - 1) / thr, thr, 0, stream>>>(ei1, ewt, deg, E);
        dinv_diag1 <<<(N * 32 + thr - 1) / thr, thr, 0, stream>>>(deg, dinv, h0, agg1, N);
        scatter1   <<<(int)(((long)E * 32 + thr - 1) / thr), thr, 0, stream>>>(
                        ei0, ei1, ewt, dinv, h0, agg1, E);
        relu_bias  <<<(N * 32 + thr - 1) / thr, thr, 0, stream>>>(agg1, b1, N);

        const int tiles = (N + 15) / 16;
        gemm_hw2_wmma<<<(tiles + 7) / 8, 256, 0, stream>>>(agg1, W2, hw2, N);

        dinv_diag2 <<<(N * 16 + thr - 1) / thr, thr, 0, stream>>>(dinv, hw2, agg2, N);
        scatter2   <<<(int)(((long)E * 16 + thr - 1) / thr), thr, 0, stream>>>(
                        ei0, ei1, ewt, dinv, hw2, agg2, E);
        tanh_bias  <<<(N * 16 + thr - 1) / thr, thr, 0, stream>>>(
                        agg2, b2, out + (size_t)t * N * Z, N);
    }
}
